// MCTDN_21328807592163
// MI455X (gfx1250) — matrix operand order repaired and checked by reading, not compiled, not run on hardware
//
#include <hip/hip_runtime.h>
#include <cstddef>

#define D_MODEL 768
#define N_HEAD  12
#define N_LAYER 6
#define BATCH   8
#define SEQ     256
#define DHEAD   64
#define MROWS   (BATCH * SEQ)                 // 2048
#define BSD     ((size_t)MROWS * D_MODEL)     // 1,572,864 floats

typedef __attribute__((ext_vector_type(16))) __bf16 v16bf;
typedef __attribute__((ext_vector_type(8)))  __bf16 v8bf;
typedef __attribute__((ext_vector_type(8)))  float  v8f;

// ---- WMMA bf16 fragment index maps (CDNA5 ISA 7.12.2) ----
// A 16x32 bf16: lane m = lane&15;  k(e) = (e>>3)*16 + (lane>>4)*8 + (e&7)
//   -> e=0..7 and e=8..15 are each 8 contiguous k (16B chunks, 32B apart)
// B 32x16 bf16: lane n = lane&15;  k(e) = (e>>3)*16 + (lane>>4)*8 + (e&7) (same as A)
//   -> e=0..7 and e=8..15 are each 8 contiguous k (16 k apart)

__device__ __forceinline__ v16bf combine8(v8bf a, v8bf b) {
  return __builtin_shufflevector(a, b, 0, 1, 2, 3, 4, 5, 6, 7,
                                 8, 9, 10, 11, 12, 13, 14, 15);
}

__device__ __forceinline__ v8f wmma_bf16(v16bf a, v16bf b, v8f c) {
#if defined(__gfx1250__)
  return __builtin_amdgcn_wmma_f32_16x16x32_bf16(false, a, false, b, (short)0, c,
                                                 false, false);
#else
  return c;
#endif
}

// =====================================================================
// GEMM: C[M,N] = act(A[M,K] @ W[K,N] + bias[N]);  M = 2048 fixed.
// Compile-time N,K,ACT -> all strided addresses are immediate offsets.
// Block: 128 threads (4 waves). Tile: 128(M) x 64(N), K step 32.
// Each wave computes 32x64 (2 A-frags x 4 shared B-frags = 8 WMMA/step).
// Software-pipelined: double-buffered LDS, next tile's global loads are
// issued before the current tile's WMMAs; ONE barrier per K-step.
//  sA: natural [m][k] bf16 (A-fragment gathers: 2x contiguous 16B)
//  sB: fragment order nt*512 + lane*16 + e (gathers: contiguous 32B)
// =====================================================================
template <int N, int K, int ACT>
__global__ __launch_bounds__(128) void gemm_kernel_t(
    const float* __restrict__ A, const float* __restrict__ W,
    const float* __restrict__ bias, float* __restrict__ C) {
  __shared__ alignas(16) __bf16 sA[2][128 * 32];       // 2 x 8 KB
  __shared__ alignas(16) __bf16 sB[2][4 * 32 * 16];    // 2 x 4 KB, fragment order
  const int t    = threadIdx.x;
  const int lane = t & 31;
  const int wave = t >> 5;          // 0..3
  const int m0   = blockIdx.y * 128;
  const int n0   = blockIdx.x * 64;
  const int row  = lane & 15;
  const int hi   = lane >> 4;

  // ---- staging maps (constant per thread) ----
  const float* aptr[4];
  int adst[4];
#pragma unroll
  for (int i = 0; i < 4; ++i) {
    int g = t + i * 128;            // 0..511 groups of 8 consecutive k
    aptr[i] = A + (size_t)(m0 + (g >> 2)) * K + ((g & 3) << 3);
    adst[i] = (g >> 2) * 32 + ((g & 3) << 3);
  }
  const float* wptr[2];
  int bdst[2];
#pragma unroll
  for (int i = 0; i < 2; ++i) {
    int g = t + i * 128;
    int c = g & 63, rg = g >> 6;    // rg 0..3
    wptr[i] = W + (size_t)(rg * 8) * N + n0 + c;
    // fragment order: nt=c>>4, lane'=((rg&1)<<4)|(c&15), e0=(rg>>1)*8
    bdst[i] = ((c >> 4) << 9) + ((((rg & 1) << 4) | (c & 15)) << 4) + ((rg >> 1) << 3);
  }

  // register staging for the in-flight tile
  float ra[4][8];
  float rb[2][8];

  // ---- prologue: load + store tile 0 into buffer 0 ----
#pragma unroll
  for (int i = 0; i < 4; ++i) {
#pragma unroll
    for (int j = 0; j < 8; ++j) ra[i][j] = aptr[i][j];
    aptr[i] += 32;
  }
#pragma unroll
  for (int i = 0; i < 2; ++i) {
#pragma unroll
    for (int j = 0; j < 8; ++j) rb[i][j] = wptr[i][(size_t)j * N];
    wptr[i] += (size_t)32 * N;
  }
#pragma unroll
  for (int i = 0; i < 4; ++i) {
    v8bf pk;
#pragma unroll
    for (int j = 0; j < 8; ++j) pk[j] = (__bf16)ra[i][j];
    *(v8bf*)&sA[0][adst[i]] = pk;
  }
#pragma unroll
  for (int i = 0; i < 2; ++i) {
    v8bf pk;
#pragma unroll
    for (int j = 0; j < 8; ++j) pk[j] = (__bf16)rb[i][j];
    *(v8bf*)&sB[0][bdst[i]] = pk;
  }

  constexpr int NT = K / 32;
  v8f acc[2][4] = {};
#pragma unroll 1
  for (int it = 0; it < NT; ++it) {
    const int cur = it & 1;
    const int nxt = cur ^ 1;
    // issue next tile's global loads (results consumed after the WMMAs)
    if (it + 1 < NT) {
#pragma unroll
      for (int i = 0; i < 4; ++i) {
#pragma unroll
        for (int j = 0; j < 8; ++j) ra[i][j] = aptr[i][j];
        aptr[i] += 32;
      }
#pragma unroll
      for (int i = 0; i < 2; ++i) {
#pragma unroll
        for (int j = 0; j < 8; ++j) rb[i][j] = wptr[i][(size_t)j * N];
        wptr[i] += (size_t)32 * N;
      }
    }
    __syncthreads();   // buffer `cur` stores (from previous step) visible

    // B fragments shared across both M sub-tiles
    v16bf bv[4];
#pragma unroll
    for (int nt = 0; nt < 4; ++nt) {
      const __bf16* bp = &sB[cur][nt * 512 + lane * 16];
      bv[nt] = combine8(*(const v8bf*)bp, *(const v8bf*)(bp + 8));
    }
    const __bf16* ap0 = &sA[cur][(wave * 32 + row) * 32 + hi * 8];
    const __bf16* ap1 = ap0 + 16 * 32;
    v16bf af0 = combine8(*(const v8bf*)ap0, *(const v8bf*)(ap0 + 16));
    v16bf af1 = combine8(*(const v8bf*)ap1, *(const v8bf*)(ap1 + 16));
#pragma unroll
    for (int nt = 0; nt < 4; ++nt) acc[0][nt] = wmma_bf16(af0, bv[nt], acc[0][nt]);
#pragma unroll
    for (int nt = 0; nt < 4; ++nt) acc[1][nt] = wmma_bf16(af1, bv[nt], acc[1][nt]);

    // convert + store next tile into the other buffer (safe: nobody reads it
    // until the barrier at the top of the next iteration)
    if (it + 1 < NT) {
#pragma unroll
      for (int i = 0; i < 4; ++i) {
        v8bf pk;
#pragma unroll
        for (int j = 0; j < 8; ++j) pk[j] = (__bf16)ra[i][j];
        *(v8bf*)&sA[nxt][adst[i]] = pk;
      }
#pragma unroll
      for (int i = 0; i < 2; ++i) {
        v8bf pk;
#pragma unroll
        for (int j = 0; j < 8; ++j) pk[j] = (__bf16)rb[i][j];
        *(v8bf*)&sB[nxt][bdst[i]] = pk;
      }
    }
  }

  // C layout: lane holds rows rbase+r (r=0..7), col = lane&15 per n-tile
  const int rbase = hi << 3;
#pragma unroll
  for (int mi = 0; mi < 2; ++mi)
#pragma unroll
    for (int nt = 0; nt < 4; ++nt)
#pragma unroll
      for (int r = 0; r < 8; ++r) {
        int grow = m0 + wave * 32 + mi * 16 + rbase + r;
        int gcol = n0 + nt * 16 + row;
        float v = acc[mi][nt][r] + bias[gcol];
        if (ACT) v = 0.5f * v * (1.0f + erff(v * 0.70710678118654752f));  // exact GELU
        C[(size_t)grow * N + gcol] = v;
      }
}

// =====================================================================
// Attention: block = (b, h, 64 q-rows); 4 waves x 16 q-rows.
// Q/K/V in [B,S,H*DH] f32.
//  sKP: K as bf16 [s][d] (contig. B-frag gathers); reused for P after scores
//  sVt: V transposed bf16 [d][s] (contig. B-frag gathers for P@V)
// =====================================================================
__global__ __launch_bounds__(128) void attn_kernel(
    const float* __restrict__ Q, const float* __restrict__ Kin,
    const float* __restrict__ V, const int* __restrict__ mask,
    float* __restrict__ O) {
  __shared__ alignas(16) __bf16 sKP[SEQ * DHEAD];   // 32 KB
  __shared__ alignas(16) __bf16 sVt[DHEAD * SEQ];   // 32 KB
  const int t    = threadIdx.x;
  const int lane = t & 31;
  const int wave = t >> 5;
  const int b    = blockIdx.x / N_HEAD;
  const int h    = blockIdx.x % N_HEAD;
  const int q0   = blockIdx.y * 64;
  const size_t base = (size_t)b * SEQ * D_MODEL + (size_t)h * DHEAD;
  const float* Kp = Kin + base;
  const float* Vp = V + base;
  const float* Qp = Q + base + (size_t)q0 * D_MODEL;

  // ---- stage K [s][d]: groups of 8 consecutive d ----
  {
    int c8 = (t & 7) << 3;     // 0..56
    int r0 = t >> 3;           // 0..15
    for (int rr = r0; rr < SEQ; rr += 16) {
      v8bf pk;
#pragma unroll
      for (int j = 0; j < 8; ++j) pk[j] = (__bf16)Kp[(size_t)rr * D_MODEL + c8 + j];
      *(v8bf*)&sKP[rr * DHEAD + c8] = pk;
    }
  }
  // ---- stage V transposed [d][s]: groups of 8 consecutive s ----
  {
    int c   = t & 63;          // d
    int rg0 = t >> 6;          // 0..1
    for (int rg = rg0; rg < 32; rg += 2) {
      v8bf pk;
#pragma unroll
      for (int j = 0; j < 8; ++j) pk[j] = (__bf16)Vp[(size_t)(rg * 8 + j) * D_MODEL + c];
      *(v8bf*)&sVt[c * SEQ + rg * 8] = pk;
    }
  }
  __syncthreads();

  const int row   = lane & 15;
  const int col0  = lane & 15;
  const int hi    = lane >> 4;
  const int rbase = hi << 3;

  // Q fragments (A-matrix 16x64 -> two 16x32 chunks); one-time global gather
  v16bf qf[2];
#pragma unroll
  for (int kc = 0; kc < 2; ++kc) {
    const float* qp = &Qp[(size_t)(wave * 16 + row) * D_MODEL + kc * 32 + hi * 8];
    v8bf lo, hi8;
#pragma unroll
    for (int j = 0; j < 8; ++j) lo[j] = (__bf16)qp[j];
#pragma unroll
    for (int j = 0; j < 8; ++j) hi8[j] = (__bf16)qp[16 + j];
    qf[kc] = combine8(lo, hi8);
  }

  float pen[16];
#pragma unroll
  for (int kt = 0; kt < 16; ++kt)
    pen[kt] = -10000.0f * (1.0f - (float)mask[b * SEQ + kt * 16 + col0]);

  // scores: 16 key tiles, all of S=256 in registers
  v8f sc[16];
#pragma unroll
  for (int kt = 0; kt < 16; ++kt) {
    v8f a = {};
#pragma unroll
    for (int kc = 0; kc < 2; ++kc) {
      const __bf16* kp = &sKP[(kt * 16 + row) * DHEAD + kc * 32 + hi * 8];
      v16bf kf = combine8(*(const v8bf*)kp, *(const v8bf*)(kp + 16));
      a = wmma_bf16(qf[kc], kf, a);
    }
    sc[kt] = a;
  }
  __syncthreads();   // all waves done reading K: sKP reusable for P

#pragma unroll
  for (int kt = 0; kt < 16; ++kt)
#pragma unroll
    for (int r = 0; r < 8; ++r)
      sc[kt][r] = sc[kt][r] * 0.125f + pen[kt];   // 1/sqrt(64), then mask

  // softmax per q-row (row rbase+r lives in one 16-lane half; xor<16 stays inside)
#pragma unroll
  for (int r = 0; r < 8; ++r) {
    float mx = -3.0e38f;
#pragma unroll
    for (int kt = 0; kt < 16; ++kt) mx = fmaxf(mx, sc[kt][r]);
#pragma unroll
    for (int o = 1; o < 16; o <<= 1) mx = fmaxf(mx, __shfl_xor(mx, o, 32));
    float sum = 0.0f;
#pragma unroll
    for (int kt = 0; kt < 16; ++kt) {
      float ev = __expf(sc[kt][r] - mx);
      sc[kt][r] = ev;
      sum += ev;
    }
#pragma unroll
    for (int o = 1; o < 16; o <<= 1) sum += __shfl_xor(sum, o, 32);
    float inv = 1.0f / sum;
#pragma unroll
    for (int kt = 0; kt < 16; ++kt) sc[kt][r] *= inv;
  }

  // store P (16x256 per wave) into reused K buffer
  __bf16* sP = sKP + wave * 16 * SEQ;
#pragma unroll
  for (int kt = 0; kt < 16; ++kt)
#pragma unroll
    for (int r = 0; r < 8; ++r)
      sP[(rbase + r) * SEQ + kt * 16 + col0] = (__bf16)sc[kt][r];
  __syncthreads();

  // O = P @ V : K-dim 256 -> 8 chunks of 32; N = 64 -> 4 tiles
  v8f oacc[4] = {};
#pragma unroll
  for (int kc = 0; kc < 8; ++kc) {
    const __bf16* pp = &sP[row * SEQ + kc * 32 + hi * 8];
    v16bf pf = combine8(*(const v8bf*)pp, *(const v8bf*)(pp + 16));
#pragma unroll
    for (int nt = 0; nt < 4; ++nt) {
      const __bf16* vp = &sVt[(nt * 16 + col0) * SEQ + kc * 32 + hi * 8];
      v16bf vf = combine8(*(const v8bf*)vp, *(const v8bf*)(vp + 16));
      oacc[nt] = wmma_bf16(pf, vf, oacc[nt]);
    }
  }

  float* Op = O + base + (size_t)q0 * D_MODEL;
#pragma unroll
  for (int nt = 0; nt < 4; ++nt)
#pragma unroll
    for (int r = 0; r < 8; ++r)
      Op[(size_t)(wave * 16 + rbase + r) * D_MODEL + nt * 16 + col0] = oacc[nt][r];
}

// =====================================================================
// out[row] = LayerNorm(resid[row] + delta[row]) * g + b ; D = 768, 256 thr
// =====================================================================
__global__ __launch_bounds__(256) void add_ln_kernel(
    const float* __restrict__ resid, const float* __restrict__ delta,
    const float* __restrict__ g, const float* __restrict__ be,
    float* __restrict__ out) {
  __shared__ float red[8];
  const int t = threadIdx.x, lane = t & 31, wave = t >> 5;
  const size_t base = (size_t)blockIdx.x * D_MODEL;
  float v0 = resid[base + t]       + delta[base + t];
  float v1 = resid[base + t + 256] + delta[base + t + 256];
  float v2 = resid[base + t + 512] + delta[base + t + 512];
  float s = v0 + v1 + v2;
#pragma unroll
  for (int o = 16; o > 0; o >>= 1) s += __shfl_xor(s, o, 32);
  if (lane == 0) red[wave] = s;
  __syncthreads();
  float tot = 0.0f;
#pragma unroll
  for (int w = 0; w < 8; ++w) tot += red[w];
  const float mean = tot * (1.0f / D_MODEL);
  __syncthreads();
  float d0 = v0 - mean, d1 = v1 - mean, d2 = v2 - mean;
  float s2 = d0 * d0 + d1 * d1 + d2 * d2;
#pragma unroll
  for (int o = 16; o > 0; o >>= 1) s2 += __shfl_xor(s2, o, 32);
  if (lane == 0) red[wave] = s2;
  __syncthreads();
  float tv = 0.0f;
#pragma unroll
  for (int w = 0; w < 8; ++w) tv += red[w];
  const float rstd = rsqrtf(tv * (1.0f / D_MODEL) + 1e-12f);
  out[base + t]       = d0 * rstd * g[t]       + be[t];
  out[base + t + 256] = d1 * rstd * g[t + 256] + be[t + 256];
  out[base + t + 512] = d2 * rstd * g[t + 512] + be[t + 512];
}

__global__ void copy2_kernel(const float* __restrict__ x, const float* __restrict__ y,
                             float* __restrict__ out, int n) {
  int i = blockIdx.x * blockDim.x + threadIdx.x;
  if (i < n) { out[i] = x[i]; out[(size_t)n + i] = y[i]; }
}

// =====================================================================
extern "C" void kernel_launch(void* const* d_in, const int* in_sizes, int n_in,
                              void* d_out, int out_size, void* d_ws, size_t ws_size,
                              hipStream_t stream) {
  (void)in_sizes; (void)n_in; (void)out_size; (void)ws_size;
  const float* x      = (const float*)d_in[0];
  const float* y      = (const float*)d_in[1];
  const int*   x_mask = (const int*)d_in[2];
  const int*   y_mask = (const int*)d_in[3];
  const float* ax_w   = (const float*)d_in[4];
  const float* ax_b   = (const float*)d_in[5];
  const float* cx_w   = (const float*)d_in[6];
  const float* cx_b   = (const float*)d_in[7];
  const float* fx_w1  = (const float*)d_in[8];
  const float* fx_b1  = (const float*)d_in[9];
  const float* fx_w2  = (const float*)d_in[10];
  const float* fx_b2  = (const float*)d_in[11];
  const float* ay_w   = (const float*)d_in[12];
  const float* ay_b   = (const float*)d_in[13];
  const float* cy_w   = (const float*)d_in[14];
  const float* cy_b   = (const float*)d_in[15];
  const float* fy_w1  = (const float*)d_in[16];
  const float* fy_b1  = (const float*)d_in[17];
  const float* fy_w2  = (const float*)d_in[18];
  const float* fy_b2  = (const float*)d_in[19];
  const float* lnx_g  = (const float*)d_in[20];
  const float* lnx_b  = (const float*)d_in[21];
  const float* lny_g  = (const float*)d_in[22];
  const float* lny_b  = (const float*)d_in[23];

  float* out = (float*)d_out;
  float* xc = out;         // resident x state, becomes output[0]
  float* yc = out + BSD;   // resident y state, becomes output[1]

  float* ws  = (float*)d_ws;
  float* qb  = ws;              // [B,S,D]
  float* kb  = ws + BSD;        // [B,S,D]
  float* vb  = ws + 2 * BSD;    // [B,S,D]
  float* mid = ws;              // [B,S,4D] (overlaps q/k/v; disjoint phase)
  float* tmp = ws + 4 * BSD;    // [B,S,D]

  copy2_kernel<<<(int)((BSD + 255) / 256), 256, 0, stream>>>(x, y, out, (int)BSD);

  // GEMM variants (N, K, ACT) with M = 2048, tile 128x64, block 128
  auto gemm_dd = [&](const float* A, const float* W, const float* bias, float* Cc) {
    gemm_kernel_t<768, 768, 0><<<dim3(12, 16), 128, 0, stream>>>(A, W, bias, Cc);
  };
  auto gemm_ffn1 = [&](const float* A, const float* W, const float* bias, float* Cc) {
    gemm_kernel_t<3072, 768, 1><<<dim3(48, 16), 128, 0, stream>>>(A, W, bias, Cc);
  };
  auto gemm_ffn2 = [&](const float* A, const float* W, const float* bias, float* Cc) {
    gemm_kernel_t<768, 3072, 0><<<dim3(12, 16), 128, 0, stream>>>(A, W, bias, Cc);
  };

  auto attn_block = [&](const float* q_in, const float* kv_in,
                        const float* w, const float* bw, const int* mask,
                        const float* g, const float* be, float* state) {
    const size_t DD = (size_t)D_MODEL * D_MODEL;
    gemm_dd(q_in,  w + 0 * DD, bw + 0 * D_MODEL, qb);
    gemm_dd(kv_in, w + 1 * DD, bw + 1 * D_MODEL, kb);
    gemm_dd(kv_in, w + 2 * DD, bw + 2 * D_MODEL, vb);
    attn_kernel<<<dim3(BATCH * N_HEAD, SEQ / 64), 128, 0, stream>>>(qb, kb, vb, mask, tmp);
    add_ln_kernel<<<MROWS, 256, 0, stream>>>(state, tmp, g, be, state);
  };

  auto ffn_block = [&](const float* w1, const float* b1, const float* w2, const float* b2,
                       const float* g, const float* be, float* state) {
    gemm_ffn1(state, w1, b1, mid);   // GELU epilogue
    gemm_ffn2(mid, w2, b2, tmp);
    add_ln_kernel<<<MROWS, 256, 0, stream>>>(state, tmp, g, be, state);
  };

  const size_t WB3 = (size_t)3 * D_MODEL * D_MODEL;
  const size_t BB3 = (size_t)3 * D_MODEL;
  const size_t W1S = (size_t)D_MODEL * 4 * D_MODEL;
  const size_t B1S = (size_t)4 * D_MODEL;
  const size_t W2S = (size_t)4 * D_MODEL * D_MODEL;
  const size_t B2S = (size_t)D_MODEL;

  for (int l = 0; l < N_LAYER; ++l) {
    // self-attention (x then y)
    attn_block(xc, xc, ax_w + l * WB3, ax_b + l * BB3, x_mask,
               lnx_g + 0, lnx_b + 0, xc);
    attn_block(yc, yc, ay_w + l * WB3, ay_b + l * BB3, y_mask,
               lny_g + 0, lny_b + 0, yc);
    // cross-attention: x queries y; then y queries the UPDATED x
    attn_block(xc, yc, cx_w + l * WB3, cx_b + l * BB3, y_mask,
               lnx_g + D_MODEL, lnx_b + D_MODEL, xc);
    attn_block(yc, xc, cy_w + l * WB3, cy_b + l * BB3, x_mask,
               lny_g + D_MODEL, lny_b + D_MODEL, yc);
    // FFN
    ffn_block(fx_w1 + l * W1S, fx_b1 + l * B1S, fx_w2 + l * W2S, fx_b2 + l * B2S,
              lnx_g + 2 * D_MODEL, lnx_b + 2 * D_MODEL, xc);
    ffn_block(fy_w1 + l * W1S, fy_b1 + l * B1S, fy_w2 + l * W2S, fy_b2 + l * B2S,
              lny_g + 2 * D_MODEL, lny_b + 2 * D_MODEL, yc);
  }
}
